// ReverseContrastiveLoss_74594991997257
// MI455X (gfx1250) — matrix operand order repaired and checked by reading, not compiled, not run on hardware
//
#include <hip/hip_runtime.h>
#include <math.h>

#define HW      65536     // 256*256 downsampled positions
#define NCLS    19
#define CAL     100
#define MAXP    684       // 2 * 19 * 18 max pairs
#define INV_TEMP 0.1f

typedef __attribute__((ext_vector_type(16))) _Float16 v16h;
typedef __attribute__((ext_vector_type(8)))  float    v8f;

struct Meta { int pairCount; int crossLen; float lossAcc; };

// ---------------- K1: pred (argmax+1), pw (max softmax prob), cat=(pred*32+gt) ----
__global__ void k_pred(const float* __restrict__ cls, const int* __restrict__ label,
                       float* __restrict__ pw, unsigned short* __restrict__ cat)
{
    int idx = blockIdx.x * blockDim.x + threadIdx.x;
    if (idx >= 2 * HW) return;
    int b = idx >> 16;
    int t = idx & (HW - 1);
    int h = t >> 8, w = t & 255;
    long src = (long)b * NCLS * 262144 + (long)(2 * h) * 512 + (2 * w);
    float vals[NCLS];
    float m = -INFINITY; int am = 0;
    #pragma unroll
    for (int c = 0; c < NCLS; ++c) {
        float v = cls[src + (long)c * 262144];
        vals[c] = v;
        if (v > m) { m = v; am = c; }
    }
    float s = 0.f;
    #pragma unroll
    for (int c = 0; c < NCLS; ++c) s += expf(vals[c] - m);
    pw[idx] = 1.0f / s;                      // max prob of softmax
    int g = label[(long)b * 262144 + (long)(2 * h) * 512 + (2 * w)] + 1;
    cat[idx] = (unsigned short)((am + 1) * 32 + g);
}

// ---------------- K2: per-category ordered index lists (first min(n,100)) ----------
// One wave per (b, p, g) category; ballot+popcount preserves ascending order.
__global__ void k_collect(const unsigned short* __restrict__ cat,
                          int* __restrict__ catIdx, int* __restrict__ catCnt)
{
    int c = blockIdx.x;                 // 0..721
    int b = c / 361, r = c % 361;
    int p = r / NCLS + 1, g = r % NCLS + 1;
    unsigned short mycat = (unsigned short)(p * 32 + g);
    int lane = threadIdx.x;
    const unsigned short* catb = cat + b * HW;
    int slot = (b * 20 + p) * 20 + g;
    int* lst = catIdx + slot * CAL;
    int base = 0;
    unsigned long long below = (1ull << lane) - 1ull;
    for (int t0 = 0; t0 < HW; t0 += 32) {
        bool match = (catb[t0 + lane] == mycat);
        unsigned long long mask = __ballot(match);
        if (match) {
            int rank = base + (int)__popcll(mask & below);
            if (rank < CAL) lst[rank] = t0 + lane;
        }
        base += (int)__popcll(mask);
    }
    if (lane == 0) catCnt[slot] = base;
}

// ---------------- K3: enumerate pairs in reference loop order ----------------------
__global__ void k_pairs(const int* __restrict__ catCnt, int* __restrict__ pairs,
                        Meta* __restrict__ meta)
{
    if (threadIdx.x != 0 || blockIdx.x != 0) return;
    int cnt = 0; long long clen = 0;
    for (int b = 0; b < 2; ++b) {
        for (int j = 1; j <= NCLS; ++j) {
            int presj = 0;
            for (int p = 1; p <= NCLS; ++p) presj += catCnt[(b * 20 + p) * 20 + j];
            if (presj == 0) continue;                       // j not in unique(gt)
            for (int k = 1; k <= NCLS; ++k) {
                if (k == j) continue;
                int ncross = catCnt[(b * 20 + j) * 20 + k]; // pred==j, gt==k
                int nphi2  = catCnt[(b * 20 + k) * 20 + k]; // pred==gt==k
                if (ncross == 0 || nphi2 == 0) continue;
                if (cnt < MAXP) {
                    pairs[cnt * 3 + 0] = b;
                    pairs[cnt * 3 + 1] = j;
                    pairs[cnt * 3 + 2] = k;
                }
                clen += ncross; cnt++;
            }
        }
    }
    meta->pairCount = cnt;
    meta->crossLen  = (int)clen;
    meta->lossAcc   = 0.f;
}

// ---------------- K4: WMMA loss ----------------------------------------------------
// A fragment: lanes 0-15 hold K=0..15, lanes 16-31 K=16..31 (matches B)
__device__ __forceinline__ v16h loadA(const _Float16 (*crossH)[256], int lane, int kk) {
    int M = lane & 15;
    int klo = (lane < 16) ? 0 : 16;
    v16h a;
    #pragma unroll
    for (int e = 0; e < 16; ++e) {
        int kq = klo + e;
        a[e] = crossH[M][kk * 32 + kq];
    }
    return a;
}
// B fragment: B = feat^T (32x16), lanes 0-15 hold K=0..15, lanes 16-31 K=16..31
__device__ __forceinline__ v16h loadB(const _Float16 (*featH)[256], int lane, int nt, int kk) {
    int N = lane & 15;
    int row = nt * 16 + N;
    if (row >= CAL) row = 0;            // clamped; masked at accumulation
    int klo = (lane < 16) ? 0 : 16;
    v16h b;
    #pragma unroll
    for (int e = 0; e < 16; ++e) b[e] = featH[row][kk * 32 + klo + e];
    return b;
}

__device__ __forceinline__ void do_phase(
        const float* __restrict__ cslb, const float* __restrict__ pwb,
        const int* __restrict__ lst, int nEff, bool useW,
        _Float16 (*featH)[256], float* rowInv, float* wset,
        const _Float16 (*crossH)[256], const float* crossInv,
        float* sums, int m0, int tid)
{
    int lane = tid & 31, wave = tid >> 5;
    __syncthreads();                                   // prior readers of featH done
    for (int r = tid; r < CAL; r += 128) rowInv[r] = 0.f;
    __syncthreads();
    // gather 100x256 f32 -> f16 LDS; f32 row sum-of-squares via wave reduce + ds atomic
    for (int r = 0; r < CAL; ++r) {
        int t = lst[r % nEff];
        float v1 = cslb[(long)tid * HW + t];
        float v2 = cslb[(long)(tid + 128) * HW + t];
        featH[r][tid]       = (_Float16)v1;
        featH[r][tid + 128] = (_Float16)v2;
        float partial = v1 * v1 + v2 * v2;
        #pragma unroll
        for (int off = 16; off > 0; off >>= 1)
            partial += __shfl_xor(partial, off, 32);
        if (lane == 0) atomicAdd(&rowInv[r], partial);
    }
    __syncthreads();
    for (int r = tid; r < CAL; r += 128) {
        float s = rowInv[r];
        rowInv[r] = (s > 0.f) ? 1.0f / sqrtf(s) : 0.f;
        if (useW) wset[r] = pwb[lst[r % nEff]];
    }
    __syncthreads();
    // WMMA: N padded to 8 tiles (tile 7 fully masked) so every wave statically
    // owns exactly 2 tiles -> no runtime-indexed accumulator arrays (no movrel).
    int nt0 = wave;          // tiles 0..3
    int nt1 = wave + 4;      // tiles 4..7 (tile 7: columns 112..127, all masked)
    v8f acc0 = {}, acc1 = {};
    #pragma unroll
    for (int kk = 0; kk < 8; ++kk) {
        v16h a  = loadA(crossH, lane, kk);
        v16h b0 = loadB(featH, lane, nt0, kk);
        acc0 = __builtin_amdgcn_wmma_f32_16x16x32_f16(
            false, a, false, b0, (short)0, acc0, false, false);
        v16h b1 = loadB(featH, lane, nt1, kk);
        acc1 = __builtin_amdgcn_wmma_f32_16x16x32_f16(
            false, a, false, b1, (short)0, acc1, false, false);
    }
    int mBase = (lane >= 16) ? 8 : 0;
    int gj0 = nt0 * 16 + (lane & 15);
    int gj1 = nt1 * 16 + (lane & 15);
    #pragma unroll
    for (int v = 0; v < 8; ++v) {
        int M  = v + mBase;
        int gi = m0 + M;
        if (gi < CAL) {
            if (gj0 < CAL) {
                float d   = acc0[v] * INV_TEMP * crossInv[M] * rowInv[gj0];
                float val = expf(d);
                if (useW) val *= wset[gj0];
                atomicAdd(&sums[M], val);
            }
            if (gj1 < CAL) {
                float d   = acc1[v] * INV_TEMP * crossInv[M] * rowInv[gj1];
                float val = expf(d);
                if (useW) val *= wset[gj1];
                atomicAdd(&sums[M], val);
            }
        }
    }
}

__global__ __launch_bounds__(128) void k_loss(
    const float* __restrict__ csl, const float* __restrict__ pw,
    const int* __restrict__ catIdx, const int* __restrict__ catCnt,
    const int* __restrict__ pairs, Meta* __restrict__ meta)
{
    __shared__ _Float16 featH[CAL][256];   // pos (phase A) / neg (phase B)
    __shared__ _Float16 crossH[16][256];
    __shared__ float rowInv[CAL];
    __shared__ float wset[CAL];
    __shared__ float crossSS[16], crossInv[16];
    __shared__ float sumPos[16], sumNeg[16];

    int pairIdx = blockIdx.y;
    if (pairIdx >= meta->pairCount) return;
    int mtile = blockIdx.x;                // 0..6
    int m0 = mtile * 16;
    int tid = threadIdx.x;

    int b = pairs[pairIdx * 3 + 0], j = pairs[pairIdx * 3 + 1], k = pairs[pairIdx * 3 + 2];
    int slotCross = (b * 20 + j) * 20 + k;
    int slotPos   = (b * 20 + k) * 20 + k;
    int slotNeg   = (b * 20 + j) * 20 + j;
    int nc  = min(catCnt[slotCross], CAL);
    int npo = min(catCnt[slotPos],   CAL);
    int nnRaw = catCnt[slotNeg];
    const int* crossL = catIdx + slotCross * CAL;
    const int* posL   = catIdx + slotPos   * CAL;
    const int* negL   = (nnRaw == 0) ? crossL : (catIdx + slotNeg * CAL);
    int nn = (nnRaw == 0) ? nc : min(nnRaw, CAL);

    const float* cslb = csl + (long)b * 256 * HW;
    const float* pwb  = pw  + (long)b * HW;

    for (int i = tid; i < 16; i += 128) {
        crossSS[i] = 0.f; sumPos[i] = 0.f; sumNeg[i] = 0.f;
    }
    __syncthreads();
    // cross tile (rows m0..m0+15), f32 norms, f16 store
    for (int idx = tid; idx < 16 * 256; idx += 128) {
        int r = idx >> 8, c = idx & 255;
        int gi = m0 + r;
        float v = 0.f;
        if (gi < CAL) {
            int t = crossL[gi % nc];
            v = cslb[(long)c * HW + t];
        }
        crossH[r][c] = (_Float16)v;
        atomicAdd(&crossSS[r], v * v);
    }
    __syncthreads();
    for (int i = tid; i < 16; i += 128)
        crossInv[i] = (crossSS[i] > 0.f) ? 1.0f / sqrtf(crossSS[i]) : 0.f;

    do_phase(cslb, pwb, posL, npo, true,  featH, rowInv, wset, crossH, crossInv, sumPos, m0, tid);
    do_phase(cslb, pwb, negL, nn,  false, featH, rowInv, wset, crossH, crossInv, sumNeg, m0, tid);

    __syncthreads();
    if (tid < 16) {
        int gi = m0 + tid;
        if (gi < CAL) {
            float contrib = logf(sumNeg[tid] / sumPos[tid] + 1.0f);
            atomicAdd(&meta->lossAcc, contrib);
        }
    }
}

// ---------------- K5: final scale --------------------------------------------------
__global__ void k_final(const Meta* __restrict__ meta, float* __restrict__ out)
{
    if (threadIdx.x == 0 && blockIdx.x == 0) {
        int cnt = meta->pairCount;
        if (cnt == 0) { out[0] = 0.f; return; }
        double denom = ((double)cnt * CAL) * ((double)cnt * CAL) / (double)meta->crossLen;
        out[0] = 0.1f * (float)((double)meta->lossAcc / denom);   // LOSS_WEIGHT=0.1
    }
}

// ---------------- host launch ------------------------------------------------------
extern "C" void kernel_launch(void* const* d_in, const int* in_sizes, int n_in,
                              void* d_out, int out_size, void* d_ws, size_t ws_size,
                              hipStream_t stream)
{
    const float* cls   = (const float*)d_in[0];  // (2,19,512,512)
    const int*   label = (const int*)  d_in[1];  // (2,1,512,512)
    const float* csl   = (const float*)d_in[2];  // (2,256,256,256)

    char* ws = (char*)d_ws;
    size_t o = 0;
    float* pw = (float*)(ws + o);                 o += (size_t)2 * HW * sizeof(float);
    unsigned short* cat = (unsigned short*)(ws + o); o += (size_t)2 * HW * sizeof(unsigned short);
    o = (o + 255) & ~(size_t)255;
    int* catIdx = (int*)(ws + o);                 o += (size_t)2 * 400 * CAL * sizeof(int);
    int* catCnt = (int*)(ws + o);                 o += (size_t)2 * 400 * sizeof(int);
    int* pairs  = (int*)(ws + o);                 o += (size_t)MAXP * 3 * sizeof(int);
    Meta* meta  = (Meta*)(ws + o);

    k_pred   <<<(2 * HW + 255) / 256, 256, 0, stream>>>(cls, label, pw, cat);
    k_collect<<<722, 32, 0, stream>>>(cat, catIdx, catCnt);
    k_pairs  <<<1, 1, 0, stream>>>(catCnt, pairs, meta);
    dim3 grid(7, MAXP);
    k_loss   <<<grid, 128, 0, stream>>>(csl, pw, catIdx, catCnt, pairs, meta);
    k_final  <<<1, 1, 0, stream>>>(meta, (float*)d_out);
}
